// HS_MSA_13941463843058
// MI455X (gfx1250) — matrix operand order repaired and checked by reading, not compiled, not run on hardware
//
#include <hip/hip_runtime.h>

// ---------------------------------------------------------------------------
// HS-MSA on MI455X (gfx1250): bf16 WMMA implicit-GEMM convs + window attention
// ---------------------------------------------------------------------------
static constexpr int BB    = 8;
static constexpr int HH    = 144;
static constexpr int WWID  = 160;
static constexpr int NWIN  = 90;          // 9*10 windows
static constexpr int WNW   = 10;
static constexpr int PIX   = BB * HH * WWID;   // 184320
static constexpr float QSCALE = 0.25f;         // (64/4)^-0.5
static constexpr float EPS    = 1e-5f;

typedef __bf16 bf16;
typedef __attribute__((ext_vector_type(16))) __bf16 bf16x16;
typedef __attribute__((ext_vector_type(8)))  __bf16 bf16x8;
typedef __attribute__((ext_vector_type(8)))  float  f32x8;

#define DEV static __device__ __forceinline__

DEV f32x8 wmma_bf16(bf16x16 a, bf16x16 b, f32x8 c) {
  return __builtin_amdgcn_wmma_f32_16x16x32_bf16(false, a, false, b, (short)0, c,
                                                 false, false);
}

// A fragment (16x32 bf16): lane half 0 -> K[0..7],[16..23]; half 1 -> K[8..15],[24..31]
DEV bf16x16 load_a(const bf16* row, int k0, int hf) {
  bf16x8 lo = *reinterpret_cast<const bf16x8*>(row + k0 + 8 * hf);
  bf16x8 hi = *reinterpret_cast<const bf16x8*>(row + k0 + 16 + 8 * hf);
  bf16x16 r;
#pragma unroll
  for (int i = 0; i < 8; ++i) { r[i] = lo[i]; r[i + 8] = hi[i]; }
  return r;
}
// B fragment (32x16 bf16): lane holds one column; half 0 -> K[0..7],[16..23]; half 1 -> K[8..15],[24..31]
DEV bf16x16 load_b(const bf16* row, int k0, int hf) {
  bf16x8 lo = *reinterpret_cast<const bf16x8*>(row + k0 + 8 * hf);
  bf16x8 hi = *reinterpret_cast<const bf16x8*>(row + k0 + 16 + 8 * hf);
  bf16x16 r;
#pragma unroll
  for (int i = 0; i < 8; ++i) { r[i] = lo[i]; r[i + 8] = hi[i]; }
  return r;
}

// ---------------------------------------------------------------------------
// init: zero BN stats + vt padding columns (windows 90..95)
// ---------------------------------------------------------------------------
__global__ void k_init(float* stats, bf16* vt) {
  int idx = blockIdx.x * 256 + threadIdx.x;
  if (idx < 256) stats[idx] = 0.0f;
  if (idx < 8 * 4 * 4096 * 6) {
    int j = 90 + idx % 6;
    int rest = idx / 6;                 // (b*4+h)*4096 + dd
    vt[rest * 96 + j] = (bf16)0.0f;
  }
}

// fuse x (NHWC) + pmt (NCHW) -> bf16 NHWC [b][h][w][128]
__global__ void k_fuse(const float* __restrict__ x, const float* __restrict__ pmt,
                       bf16* __restrict__ fused) {
  int idx = blockIdx.x * 256 + threadIdx.x;
  if (idx >= PIX * 128) return;
  int ch = idx & 127, p = idx >> 7;
  int b = p / (HH * WWID), rem = p % (HH * WWID);
  float v = (ch < 64) ? x[p * 64 + ch]
                      : pmt[(b * 64 + (ch - 64)) * (HH * WWID) + rem];
  fused[idx] = (bf16)v;
}

// weight prep: conv weights -> [cout][9*cin] bf16, proj weights -> bf16 row-major
__global__ void k_prepw(const float* __restrict__ c1w, const float* __restrict__ c2w,
                        const float* __restrict__ Wq, const float* __restrict__ Wk,
                        const float* __restrict__ Wv, const float* __restrict__ Wo,
                        bf16* __restrict__ w1t, bf16* __restrict__ w2t,
                        bf16* __restrict__ wqb, bf16* __restrict__ wkb,
                        bf16* __restrict__ wvb, bf16* __restrict__ wob) {
  int idx = blockIdx.x * 256 + threadIdx.x;
  if (idx < 64 * 1152) {
    int co = idx / 1152, k = idx % 1152, tap = k / 128, ci = k % 128;
    w1t[idx] = (bf16)c1w[(co * 128 + ci) * 9 + tap];
  } else if (idx < 64 * 1152 + 64 * 576) {
    int i = idx - 64 * 1152;
    int co = i / 576, k = i % 576, tap = k / 64, ci = k % 64;
    w2t[i] = (bf16)c2w[(co * 64 + ci) * 9 + tap];
  } else if (idx < 64 * 1152 + 64 * 576 + 4 * 4096) {
    int i = idx - (64 * 1152 + 64 * 576);
    int m = i / 4096, e = i % 4096;
    const float* src = (m == 0) ? Wq : (m == 1) ? Wk : (m == 2) ? Wv : Wo;
    bf16* dst = (m == 0) ? wqb : (m == 1) ? wkb : (m == 2) ? wvb : wob;
    dst[e] = (bf16)src[e];
  }
}

// ---------------------------------------------------------------------------
// implicit-GEMM conv3x3 (pad 1) + BN batch-stat accumulation
// block: 128 threads (4 waves), 16 output pixels, wave w -> couts [16w,16w+16)
// ---------------------------------------------------------------------------
template <int CIN>
__global__ __launch_bounds__(128) void k_conv(const bf16* __restrict__ in,
                                              const bf16* __restrict__ wT,
                                              float* __restrict__ out,
                                              float* __restrict__ gSum,
                                              float* __restrict__ gSq) {
  constexpr int K = 9 * CIN;
  __shared__ float redS[64], redQ[64];
  int tid = threadIdx.x;
  int wave = tid >> 5, lane = tid & 31, hf = lane >> 4, ml = lane & 15;
  int p0 = blockIdx.x * 16;
  int pm = p0 + ml;
  int b = pm / (HH * WWID), rem = pm % (HH * WWID);
  int hh = rem / WWID, ww = rem % WWID;
  int co = wave * 16 + ml;
  const bf16* brow = wT + co * K;
  f32x8 acc = {};
#pragma unroll
  for (int tap = 0; tap < 9; ++tap) {
    int h2 = hh + tap / 3 - 1;
    int w2 = ww + tap % 3 - 1;
    bool valid = ((unsigned)h2 < (unsigned)HH) && ((unsigned)w2 < (unsigned)WWID);
    const bf16* arow = in + ((b * HH + h2) * WWID + w2) * CIN;
#pragma unroll
    for (int kc = 0; kc < CIN / 32; ++kc) {
      bf16x16 a;
      if (valid) a = load_a(arow, kc * 32, hf);
      else       a = bf16x16{};
      bf16x16 bb = load_b(brow, tap * CIN + kc * 32, hf);
      acc = wmma_bf16(a, bb, acc);
    }
  }
  if (tid < 64) { redS[tid] = 0.0f; redQ[tid] = 0.0f; }
  __syncthreads();
  float ls = 0.0f, lq = 0.0f;
#pragma unroll
  for (int r = 0; r < 8; ++r) {
    int m = r + 8 * hf;
    float v = acc[r];
    out[(p0 + m) * 64 + co] = v;
    ls += v; lq += v * v;
  }
  atomicAdd(&redS[co], ls);
  atomicAdd(&redQ[co], lq);
  __syncthreads();
  if (tid < 64) { atomicAdd(&gSum[tid], redS[tid]); atomicAdd(&gSq[tid], redQ[tid]); }
}

// training-mode BN (biased batch stats) + ReLU, f32 -> bf16
__global__ void k_bnrelu(const float* __restrict__ in, const float* __restrict__ g,
                         const float* __restrict__ be, const float* __restrict__ gSum,
                         const float* __restrict__ gSq, bf16* __restrict__ out) {
  int idx = blockIdx.x * 256 + threadIdx.x;
  if (idx >= PIX * 64) return;
  int c = idx & 63;
  const float invn = 1.0f / (float)PIX;
  float mean = gSum[c] * invn;
  float var  = gSq[c] * invn - mean * mean;
  float sc   = g[c] * rsqrtf(var + EPS);
  float v    = (in[idx] - mean) * sc + be[c];
  out[idx] = (bf16)(v > 0.0f ? v : 0.0f);
}

// ---------------------------------------------------------------------------
// QKV projection + window scatter.
// q,k -> [b][n][c][s] (q pre-scaled); v -> per-head transposed [b][h][dd][96]
// block: 128 threads (4 waves), each wave owns 16 consecutive pixels (one window row)
// ---------------------------------------------------------------------------
__global__ __launch_bounds__(128) void k_qkv(const bf16* __restrict__ xp,
                                             const bf16* __restrict__ wq,
                                             const bf16* __restrict__ wk,
                                             const bf16* __restrict__ wv,
                                             bf16* __restrict__ qw, bf16* __restrict__ kw,
                                             bf16* __restrict__ vt) {
  int tid = threadIdx.x, wave = tid >> 5, lane = tid & 31, hf = lane >> 4, nl = lane & 15;
  int p0 = (blockIdx.x * 4 + wave) * 16;
  int b = p0 / (HH * WWID), rem = p0 % (HH * WWID);
  int hh = rem / WWID, ww0 = rem % WWID;
  int b0 = hh & 15;
  int n = (hh >> 4) * WNW + (ww0 >> 4);
  const bf16* arow = xp + (p0 + nl) * 64;        // A row m = lane&15
  bf16x16 a0 = load_a(arow, 0, hf);
  bf16x16 a1 = load_a(arow, 32, hf);
  const bf16* mats[3] = {wq, wk, wv};
#pragma unroll
  for (int m3 = 0; m3 < 3; ++m3) {
#pragma unroll
    for (int nt = 0; nt < 4; ++nt) {
      int o = nt * 16 + nl;
      const bf16* brow = mats[m3] + o * 64;
      f32x8 acc = {};
      acc = wmma_bf16(a0, load_b(brow, 0, hf), acc);
      acc = wmma_bf16(a1, load_b(brow, 32, hf), acc);
      if (m3 == 2) {
        int head = o >> 4, cl = o & 15;
#pragma unroll
        for (int r = 0; r < 8; ++r) {
          int s = b0 * 16 + r + 8 * hf;
          int dd = cl * 256 + s;
          vt[((b * 4 + head) * 4096 + dd) * 96 + n] = (bf16)acc[r];
        }
      } else {
        float sc = (m3 == 0) ? QSCALE : 1.0f;
        bf16* dst = ((m3 == 0) ? qw : kw) +
                    ((b * NWIN + n) * 64 + o) * 256 + b0 * 16 + 8 * hf;
        bf16x8 pk;
#pragma unroll
        for (int r = 0; r < 8; ++r) pk[r] = (bf16)(acc[r] * sc);
        *reinterpret_cast<bf16x8*>(dst) = pk;
      }
    }
  }
}

// ---------------------------------------------------------------------------
// scores + softmax: sim = (q.k + pos)*mask, exact softmax over 90 cols.
// block (i_tile, h, b): 192 threads = 6 waves = 6 j-tiles; K = 4096 (128 steps)
// attn written bf16 [b][h][96][96], zero padded
// ---------------------------------------------------------------------------
__global__ __launch_bounds__(192) void k_scores(const bf16* __restrict__ qw,
                                                const bf16* __restrict__ kw,
                                                const float* __restrict__ pos,
                                                const float* __restrict__ mask,
                                                bf16* __restrict__ attn) {
  __shared__ float sim[16][96];
  __shared__ float rsum[16];
  int i0 = blockIdx.x * 16, h = blockIdx.y, b = blockIdx.z;
  int tid = threadIdx.x, wave = tid >> 5, lane = tid & 31, hf = lane >> 4, nl = lane & 15;
  int i = i0 + nl;            int iq = (i < NWIN) ? i : 0;
  int j = wave * 16 + nl;     int jq = (j < NWIN) ? j : 0;
  const bf16* arow = qw + (size_t)(b * NWIN + iq) * 16384 + h * 4096;
  const bf16* brow = kw + (size_t)(b * NWIN + jq) * 16384 + h * 4096;
  f32x8 acc = {};
#pragma unroll 8
  for (int kc = 0; kc < 128; ++kc) {
    if ((kc & 15) == 0) {
      __builtin_prefetch(arow + kc * 32 + 2048, 0, 0);
      __builtin_prefetch(brow + kc * 32 + 2048, 0, 0);
    }
    acc = wmma_bf16(load_a(arow, kc * 32, hf), load_b(brow, kc * 32, hf), acc);
  }
#pragma unroll
  for (int r = 0; r < 8; ++r) {
    int row = r + 8 * hf;
    int ii = i0 + row, jj = wave * 16 + nl;
    float v = acc[r];
    if (ii < NWIN && jj < NWIN)
      v = (v + pos[(h * NWIN + ii) * NWIN + jj]) * mask[ii * NWIN + jj];
    sim[row][wave * 16 + nl] = v;
  }
  __syncthreads();
  if (tid < 16) {
    int row = tid, ii = i0 + row;
    float sum = 1.0f;
    if (ii < NWIN) {
      float mx = -1e30f;
      for (int jj = 0; jj < NWIN; ++jj) mx = fmaxf(mx, sim[row][jj]);
      sum = 0.0f;
      for (int jj = 0; jj < NWIN; ++jj) {
        float e = __expf(sim[row][jj] - mx);
        sim[row][jj] = e;
        sum += e;
      }
    }
    rsum[row] = sum;
  }
  __syncthreads();
  for (int lin = tid; lin < 16 * 96; lin += 192) {
    int row = lin / 96, jj = lin % 96;
    int ii = i0 + row;
    float a = (ii < NWIN && jj < NWIN) ? sim[row][jj] / rsum[row] : 0.0f;
    attn[((b * 4 + h) * 96 + i0 + row) * 96 + jj] = (bf16)a;
  }
}

// ---------------------------------------------------------------------------
// o = attn @ v : M=16 (i-tile), N=4096, K=96. A frags preloaded & reused.
// block (i_tile, h, b): 256 threads = 8 waves, wave strides over 256 n-tiles
// ---------------------------------------------------------------------------
__global__ __launch_bounds__(256) void k_av(const bf16* __restrict__ attn,
                                            const bf16* __restrict__ vt,
                                            bf16* __restrict__ o) {
  int i0 = blockIdx.x * 16, h = blockIdx.y, b = blockIdx.z;
  int tid = threadIdx.x, wave = tid >> 5, lane = tid & 31, hf = lane >> 4, nl = lane & 15;
  const bf16* arow = attn + ((b * 4 + h) * 96 + i0 + nl) * 96;
  bf16x16 a0 = load_a(arow, 0, hf);
  bf16x16 a1 = load_a(arow, 32, hf);
  bf16x16 a2 = load_a(arow, 64, hf);
  for (int nt = wave; nt < 256; nt += 8) {
    int dd = nt * 16 + nl;
    const bf16* brow = vt + (size_t)((b * 4 + h) * 4096 + dd) * 96;
    f32x8 acc = {};
    acc = wmma_bf16(a0, load_b(brow, 0, hf), acc);
    acc = wmma_bf16(a1, load_b(brow, 32, hf), acc);
    acc = wmma_bf16(a2, load_b(brow, 64, hf), acc);
    int c = h * 16 + (dd >> 8), s = dd & 255;
#pragma unroll
    for (int r = 0; r < 8; ++r) {
      int ii = i0 + r + 8 * hf;
      if (ii < NWIN)
        o[((size_t)(b * NWIN + ii) * 64 + c) * 256 + s] = (bf16)acc[r];
    }
  }
}

// ---------------------------------------------------------------------------
// out = unwindow(o) @ Wout.T + bout.  One block per (window, batch).
// o[c][s] tile staged transposed in LDS -> A rows contiguous; f32 NHWC output.
// ---------------------------------------------------------------------------
__global__ __launch_bounds__(256) void k_out(const bf16* __restrict__ o,
                                             const bf16* __restrict__ wo,
                                             const float* __restrict__ bout,
                                             float* __restrict__ out) {
  __shared__ bf16 lt[256][72];
  int n = blockIdx.x, b = blockIdx.y;
  int tid = threadIdx.x, wave = tid >> 5, lane = tid & 31, hf = lane >> 4, nl = lane & 15;
  const bf16* src = o + (size_t)(b * NWIN + n) * 16384;
  {
    int c = tid >> 2, sBase = (tid & 3) * 64;
#pragma unroll
    for (int ch = 0; ch < 8; ++ch) {
      bf16x8 v = *reinterpret_cast<const bf16x8*>(src + c * 256 + sBase + ch * 8);
#pragma unroll
      for (int e = 0; e < 8; ++e) lt[sBase + ch * 8 + e][c] = v[e];
    }
  }
  __syncthreads();
  int hn2 = n / WNW, wn2 = n % WNW;
  for (int mt = wave * 2; mt < wave * 2 + 2; ++mt) {
    const bf16* arow = &lt[mt * 16 + nl][0];
    bf16x16 a0 = load_a(arow, 0, hf);
    bf16x16 a1 = load_a(arow, 32, hf);
#pragma unroll
    for (int nt = 0; nt < 4; ++nt) {
      int co = nt * 16 + nl;
      const bf16* brow = wo + co * 64;
      f32x8 acc = {};
      acc = wmma_bf16(a0, load_b(brow, 0, hf), acc);
      acc = wmma_bf16(a1, load_b(brow, 32, hf), acc);
      float bo = bout[co];
#pragma unroll
      for (int r = 0; r < 8; ++r) {
        int s = mt * 16 + r + 8 * hf;
        int hh = hn2 * 16 + (s >> 4), ww = wn2 * 16 + (s & 15);
        out[((size_t)(b * HH + hh) * WWID + ww) * 64 + co] = acc[r] + bo;
      }
    }
  }
}

// ---------------------------------------------------------------------------
// workspace layout (bytes) -- intermediates aliased where lifetimes allow
// ---------------------------------------------------------------------------
static constexpr size_t SZ_FUSED = (size_t)PIX * 128 * 2;     // 47.2 MB
static constexpr size_t SZ_HALF  = (size_t)PIX * 64 * 2;      // 23.6 MB
static constexpr size_t OFF_FUSED = 0;                        // fused; later q|k
static constexpr size_t OFF_CONV  = OFF_FUSED + SZ_FUSED;     // conv out f32 (reused)
static constexpr size_t OFF_ACT1  = OFF_CONV + (size_t)PIX * 64 * 4;  // act1; later o
static constexpr size_t OFF_XPMT  = OFF_ACT1 + SZ_HALF;
static constexpr size_t OFF_VT    = OFF_XPMT + SZ_HALF;
static constexpr size_t OFF_ATTN  = OFF_VT + (size_t)8 * 4 * 4096 * 96 * 2;
static constexpr size_t OFF_W1T   = OFF_ATTN + (size_t)8 * 4 * 96 * 96 * 2;
static constexpr size_t OFF_W2T   = OFF_W1T + 64 * 1152 * 2;
static constexpr size_t OFF_WQ    = OFF_W2T + 64 * 576 * 2;
static constexpr size_t OFF_WK    = OFF_WQ + 8192;
static constexpr size_t OFF_WV    = OFF_WK + 8192;
static constexpr size_t OFF_WO    = OFF_WV + 8192;
static constexpr size_t OFF_STATS = OFF_WO + 8192;

extern "C" void kernel_launch(void* const* d_in, const int* in_sizes, int n_in,
                              void* d_out, int out_size, void* d_ws, size_t ws_size,
                              hipStream_t stream) {
  (void)in_sizes; (void)n_in; (void)out_size; (void)ws_size;
  const float* x    = (const float*)d_in[0];
  const float* pmt  = (const float*)d_in[1];
  const float* c1w  = (const float*)d_in[2];
  const float* g1   = (const float*)d_in[3];
  const float* b1   = (const float*)d_in[4];
  const float* c2w  = (const float*)d_in[5];
  const float* g2   = (const float*)d_in[6];
  const float* b2   = (const float*)d_in[7];
  const float* Wq   = (const float*)d_in[8];
  const float* Wk   = (const float*)d_in[9];
  const float* Wv   = (const float*)d_in[10];
  const float* Wo   = (const float*)d_in[11];
  const float* bo   = (const float*)d_in[12];
  const float* pos  = (const float*)d_in[13];
  const float* mask = (const float*)d_in[14];
  float* out = (float*)d_out;
  char* ws = (char*)d_ws;

  bf16*  fused = (bf16*)(ws + OFF_FUSED);
  bf16*  qw    = (bf16*)(ws + OFF_FUSED);            // alias: fused dead after conv1
  bf16*  kw    = (bf16*)(ws + OFF_FUSED + SZ_HALF);
  float* convo = (float*)(ws + OFF_CONV);
  bf16*  act1  = (bf16*)(ws + OFF_ACT1);
  bf16*  obuf  = (bf16*)(ws + OFF_ACT1);             // alias: act1 dead after conv2
  bf16*  xpmt  = (bf16*)(ws + OFF_XPMT);
  bf16*  vt    = (bf16*)(ws + OFF_VT);
  bf16*  attn  = (bf16*)(ws + OFF_ATTN);
  bf16*  w1t   = (bf16*)(ws + OFF_W1T);
  bf16*  w2t   = (bf16*)(ws + OFF_W2T);
  bf16*  wqb   = (bf16*)(ws + OFF_WQ);
  bf16*  wkb   = (bf16*)(ws + OFF_WK);
  bf16*  wvb   = (bf16*)(ws + OFF_WV);
  bf16*  wob   = (bf16*)(ws + OFF_WO);
  float* stats = (float*)(ws + OFF_STATS);           // [sum1|sq1|sum2|sq2] x64

  k_init <<<(8 * 4 * 4096 * 6 + 255) / 256, 256, 0, stream>>>(stats, vt);
  k_fuse <<<(PIX * 128 + 255) / 256, 256, 0, stream>>>(x, pmt, fused);
  k_prepw<<<(64 * 1152 + 64 * 576 + 4 * 4096 + 255) / 256, 256, 0, stream>>>(
      c1w, c2w, Wq, Wk, Wv, Wo, w1t, w2t, wqb, wkb, wvb, wob);

  k_conv<128><<<PIX / 16, 128, 0, stream>>>(fused, w1t, convo, stats + 0, stats + 64);
  k_bnrelu<<<(PIX * 64 + 255) / 256, 256, 0, stream>>>(convo, g1, b1,
                                                       stats + 0, stats + 64, act1);
  k_conv<64><<<PIX / 16, 128, 0, stream>>>(act1, w2t, convo, stats + 128, stats + 192);
  k_bnrelu<<<(PIX * 64 + 255) / 256, 256, 0, stream>>>(convo, g2, b2,
                                                       stats + 128, stats + 192, xpmt);

  k_qkv   <<<PIX / 64, 128, 0, stream>>>(xpmt, wqb, wkb, wvb, qw, kw, vt);
  k_scores<<<dim3(6, 4, 8), 192, 0, stream>>>(qw, kw, pos, mask, attn);
  k_av    <<<dim3(6, 4, 8), 256, 0, stream>>>(attn, vt, obuf);
  k_out   <<<dim3(NWIN, BB), 256, 0, stream>>>(obuf, wob, bo, out);
}
